// SymmetricQuantLinear_51694226374739
// MI455X (gfx1250) — matrix operand order repaired and checked by reading, not compiled, not run on hardware
//
#include <hip/hip_runtime.h>

typedef __attribute__((ext_vector_type(16))) _Float16 v16h;
typedef __attribute__((ext_vector_type(8)))  _Float16 v8h;
typedef __attribute__((ext_vector_type(2)))  _Float16 h2v;
typedef __attribute__((ext_vector_type(8)))  float    v8f;
typedef __attribute__((ext_vector_type(4)))  unsigned int v4u;

#if __has_builtin(__builtin_amdgcn_global_load_async_to_lds_b128)
#define USE_ASYNC_LDS 1
#else
#define USE_ASYNC_LDS 0
#endif

#if USE_ASYNC_LDS
// ROCm 7.2 clang-22 signature (from diagnostic):
//   void (int4 addrspace(1)*, int4 addrspace(3)*, imm int offset, imm int cpol)
typedef int v4i_vs __attribute__((vector_size(16)));
typedef __attribute__((address_space(1))) v4i_vs as1_v4i;
typedef __attribute__((address_space(3))) v4i_vs as3_v4i;
#endif

namespace {
constexpr int MDIM = 4096;
constexpr int KDIM = 4096;
constexpr int NDIM = 11008;
constexpr int BM = 128;
constexpr int BN = 128;
constexpr int BK = 64;
constexpr int THREADS = 256;

struct V16Pair { v8h lo, hi; };
}

__device__ __forceinline__ void wait_async_copies() {
#if USE_ASYNC_LDS
#if __has_builtin(__builtin_amdgcn_s_wait_asynccnt)
  __builtin_amdgcn_s_wait_asynccnt(0);
#else
  asm volatile("s_wait_asynccnt 0" ::: "memory");
#endif
#endif
}

__global__ __launch_bounds__(THREADS)
void q4_wmma_gemm(const _Float16* __restrict__ X,
                  const unsigned int* __restrict__ Wp,
                  const float* __restrict__ Ws,
                  float* __restrict__ Out) {
  // Double-buffered tiles: A row-major [m][k], B as [n][k] (K contiguous).
  __shared__ _Float16 As[2][BM * BK];   // 2 * 16 KB
  __shared__ _Float16 Bs[2][BN * BK];   // 2 * 16 KB

  const int tid  = threadIdx.x;
  const int lane = tid & 31;
  const int wid  = tid >> 5;
  const int wm   = wid & 3;    // wave row: 4 waves x 32 rows of M
  const int wn   = wid >> 2;   // wave col: 2 waves x 64 cols of N

  const int bm0 = blockIdx.y * BM;
  const int bn0 = blockIdx.x * BN;

  v8f acc[2][4] = {};

  // A tile staging: 256 threads x 4 chunks of 16 B (128 rows x 128 B).
  const int a_row = tid >> 3;          // 0..31
  const int a_col = (tid & 7) * 8;     // half index within BK
  // B tile staging: each thread handles 4 packed-K rows x 4 consecutive n.
  const int b_kp  = tid >> 5;          // 0..7  (packed-k row within tile)
  const int b_n   = (tid & 31) * 4;    // 0..124

  auto loadA = [&](int buf, int k0) {
#pragma unroll
    for (int r = 0; r < 4; ++r) {
      const int row = a_row + 32 * r;
      const _Float16* gsrc = X + (size_t)(bm0 + row) * KDIM + k0 + a_col;
      _Float16* ldst = &As[buf][row * BK + a_col];
#if USE_ASYNC_LDS
      // CDNA5: DMA 16 B straight into LDS, tracked by ASYNCcnt.
      __builtin_amdgcn_global_load_async_to_lds_b128(
          (as1_v4i*)gsrc, (as3_v4i*)ldst, /*offset=*/0, /*cpol=*/0);
#else
      *reinterpret_cast<v8h*>(ldst) = *reinterpret_cast<const v8h*>(gsrc);
#endif
    }
  };

  auto loadB = [&](int buf, int k0) {
    const int kp0 = k0 >> 1;  // packed row index in global W
    const h2v bias = {(_Float16)-1032.0f, (_Float16)-1032.0f};
#pragma unroll
    for (int r = 0; r < 4; ++r) {
      const int kp = b_kp + 8 * r;  // 0..31 packed rows in this tile
      const v4u p = *reinterpret_cast<const v4u*>(
          Wp + (size_t)(kp0 + kp) * NDIM + bn0 + b_n);
#pragma unroll
      for (int j = 0; j < 4; ++j) {
        const unsigned int v = p[j];
        // fp16 exponent trick: 0x6400|nib == 1024+nib ; subtract 1032 -> nib-8
        const unsigned int d = 0x64006400u | (v & 0xFu) | ((v & 0xF0u) << 12);
        const h2v hv = __builtin_bit_cast(h2v, d) + bias;   // v_pk_add_f16
        // low nibble -> even k (2*kp), high nibble -> odd k (2*kp+1)
        *reinterpret_cast<h2v*>(&Bs[buf][(b_n + j) * BK + 2 * kp]) = hv;
      }
    }
  };

  // A fragment (16x32 f16): lanes 0-15 hold M=lane, K 0-15;
  // lanes 16-31 hold K 16-31 (matches B fragment order).
  auto loadAfrag = [&](int buf, int mt, int ks) -> v16h {
    const int row = wm * 32 + mt * 16 + (lane & 15);
    const int g = lane >> 4;
    const _Float16* base = &As[buf][row * BK + ks * 32 + g * 16];
    V16Pair pr;
    pr.lo = *reinterpret_cast<const v8h*>(base);        // K g*16 .. +8
    pr.hi = *reinterpret_cast<const v8h*>(base + 8);    // K g*16+8 .. +8
    return __builtin_bit_cast(v16h, pr);
  };

  // B fragment (32x16 f16): lanes 0-15 col N=lane K 0-15, lanes 16-31 K 16-31.
  auto loadBfrag = [&](int buf, int nt, int ks) -> v16h {
    const int col = wn * 64 + nt * 16 + (lane & 15);
    const int g = lane >> 4;
    return *reinterpret_cast<const v16h*>(
        &Bs[buf][col * BK + ks * 32 + g * 16]);         // 32 B contiguous
  };

  loadA(0, 0);
  loadB(0, 0);
  wait_async_copies();
  __syncthreads();

  constexpr int NITER = KDIM / BK;  // 64
  for (int it = 0; it < NITER; ++it) {
    const int buf = it & 1;
    if (it + 1 < NITER) {           // prefetch next tile into other buffer
      loadA(buf ^ 1, (it + 1) * BK);
      loadB(buf ^ 1, (it + 1) * BK);
    }
#pragma unroll
    for (int ks = 0; ks < 2; ++ks) {
      v16h af[2], bf[4];
#pragma unroll
      for (int mt = 0; mt < 2; ++mt) af[mt] = loadAfrag(buf, mt, ks);
#pragma unroll
      for (int nt = 0; nt < 4; ++nt) bf[nt] = loadBfrag(buf, nt, ks);
#pragma unroll
      for (int mt = 0; mt < 2; ++mt)
#pragma unroll
        for (int nt = 0; nt < 4; ++nt)
          acc[mt][nt] = __builtin_amdgcn_wmma_f32_16x16x32_f16(
              false, af[mt], false, bf[nt], (short)0, acc[mt][nt],
              false, false);
    }
    wait_async_copies();   // next-buffer async LDS writes visible ...
    __syncthreads();       // ... before anyone reads them
  }

  // Epilogue: C/D layout -> VGPR r holds M=r (lanes 0-15) / M=8+r (16-31),
  // N = lane&15. Apply per-column fp32 scale, store float.
  const int col_in_tile = lane & 15;
  const int row_half = (lane >> 4) * 8;
  float s[4];
#pragma unroll
  for (int nt = 0; nt < 4; ++nt)
    s[nt] = Ws[bn0 + wn * 64 + nt * 16 + col_in_tile];

#pragma unroll
  for (int mt = 0; mt < 2; ++mt) {
#pragma unroll
    for (int nt = 0; nt < 4; ++nt) {
      const int col = bn0 + wn * 64 + nt * 16 + col_in_tile;
#pragma unroll
      for (int r = 0; r < 8; ++r) {
        const int row = bm0 + wm * 32 + mt * 16 + row_half + r;
        Out[(size_t)row * NDIM + col] = acc[mt][nt][r] * s[nt];
      }
    }
  }
}

extern "C" void kernel_launch(void* const* d_in, const int* in_sizes, int n_in,
                              void* d_out, int out_size, void* d_ws, size_t ws_size,
                              hipStream_t stream) {
  const _Float16* X       = (const _Float16*)d_in[0];
  const unsigned int* Wp  = (const unsigned int*)d_in[1];
  const float* Ws         = (const float*)d_in[2];
  float* Out              = (float*)d_out;

  dim3 grid(NDIM / BN, MDIM / BM);   // 86 x 32 blocks
  dim3 block(THREADS);
  hipLaunchKernelGGL(q4_wmma_gemm, grid, block, 0, stream, X, Wp, Ws, Out);
}
